// SwinTransformerBlock3D_84000970375764
// MI455X (gfx1250) — matrix operand order repaired and checked by reading, not compiled, not run on hardware
//
#include <hip/hip_runtime.h>
#include <cstdint>
#include <cstddef>

// ---------------------------------------------------------------------------
// Types for CDNA5 WMMA (wave32, v_wmma_f32_16x16x32_bf16)
// ---------------------------------------------------------------------------
typedef __bf16 bf16;
typedef __attribute__((ext_vector_type(16))) __bf16 v16bf;
typedef __attribute__((ext_vector_type(8)))  float  v8f;

union BF16Tile { v16bf v; uint4 q[2]; };

__device__ __forceinline__ bf16 f2b(float f) {
  union { float f; uint32_t u; } in; in.f = f;
  uint32_t u = in.u;
  u += 0x7FFFu + ((u >> 16) & 1u);          // round to nearest even
  union { uint16_t s; bf16 h; } out; out.s = (uint16_t)(u >> 16);
  return out.h;
}

__device__ __forceinline__ float wave_sum(float v) {
#pragma unroll
  for (int off = 16; off > 0; off >>= 1) v += __shfl_xor(v, off, 32);
  return v;
}

// ---------------------------------------------------------------------------
// Problem constants
// ---------------------------------------------------------------------------
constexpr int Bn = 8, Dd = 8, Hh = 28, Ww = 28, Cc = 384;
constexpr int HEADS = 12, HD = 32;
constexpr int NTOK = 98, NPAD = 112;          // 7 * 16
constexpr int NSQ = NPAD * NPAD;              // 12544
constexpr int NWIN = 512;                     // B * 64
constexpr int M_WIN = NWIN * NPAD;            // 57344 (multiple of 256)
constexpr int M_TOK = Bn * Dd * Hh * Ww;      // 50176 (multiple of 256)
constexpr int MLPH = 4 * Cc;                  // 1536
constexpr int BOT = 64;

// ---------------------------------------------------------------------------
// Weight f32 -> bf16 conversion
// ---------------------------------------------------------------------------
__global__ void f2bf_kernel(const float* __restrict__ in, bf16* __restrict__ out, int n) {
  int i = blockIdx.x * 256 + threadIdx.x;
  if (i < n) out[i] = f2b(in[i]);
}

// ---------------------------------------------------------------------------
// Precomputed attention tables (rel-pos bias per head; shift mask per window)
// ---------------------------------------------------------------------------
__device__ __forceinline__ int region_cnt(int t, int wd, int wh, int ww) {
  int zd = t / 49, rem = t % 49, yh = rem / 7, xw = rem % 7;
  int d = wd * 2 + zd, h = wh * 7 + yh, g = ww * 7 + xw;
  int rd = (d < 6) ? 0 : ((d < 7) ? 1 : 2);
  int rh = (h < 21) ? 0 : ((h < 25) ? 1 : 2);
  int rw = (g < 21) ? 0 : ((g < 25) ? 1 : 2);
  return rd * 9 + rh * 3 + rw;
}
__device__ __forceinline__ int relidx(int m, int n) {
  int zm = m / 49, rm = m % 49, ym = rm / 7, xm = rm % 7;
  int zn = n / 49, rn = n % 49, yn = rn / 7, xn = rn % 7;
  return (zm - zn + 1) * 169 + (ym - yn + 6) * 13 + (xm - xn + 6);
}

__global__ __launch_bounds__(256)
void build_bias_kernel(const float* __restrict__ rpb, float* __restrict__ biasT) {
  int idx = blockIdx.x * 256 + threadIdx.x;          // < 12 * 12544
  int head = idx / NSQ, rem = idx % NSQ, m = rem / NPAD, n = rem % NPAD;
  float v;
  if (n >= NTOK)      v = -1e30f;                    // padded key -> masked out
  else if (m >= NTOK) v = 0.0f;                      // padded query row, don't care
  else                v = rpb[relidx(m, n) * HEADS + head];
  biasT[idx] = v;
}

__global__ __launch_bounds__(256)
void build_mask_kernel(float* __restrict__ maskT) {
  int idx = blockIdx.x * 256 + threadIdx.x;          // < 64 * 12544
  int iw = idx / NSQ, rem = idx % NSQ, m = rem / NPAD, n = rem % NPAD;
  float v = 0.0f;
  if (m < NTOK && n < NTOK) {
    int wd = iw >> 4, wh = (iw >> 2) & 3, ww = iw & 3;
    v = (region_cnt(m, wd, wh, ww) != region_cnt(n, wd, wh, ww)) ? -100.0f : 0.0f;
  }
  maskT[idx] = v;
}

// ---------------------------------------------------------------------------
// LN1 + cyclic shift(-1,-3,-3) + 3D window partition -> bf16 Xw[512][112][384]
// ---------------------------------------------------------------------------
__global__ __launch_bounds__(256)
void ln1_part_kernel(const float* __restrict__ x, const float* __restrict__ g,
                     const float* __restrict__ bta, bf16* __restrict__ xw) {
  int wid = threadIdx.x >> 5, lane = threadIdx.x & 31;
  int tk = blockIdx.x * 8 + wid;              // 0 .. 57343
  int win = tk / NPAD, tok = tk % NPAD;
  bf16* orow = xw + (size_t)tk * Cc;
  if (tok >= NTOK) {
    for (int i = lane; i < Cc; i += 32) orow[i] = f2b(0.0f);
    return;
  }
  int bidx = win >> 6, iw = win & 63;
  int wd = iw >> 4, wh = (iw >> 2) & 3, wwj = iw & 3;
  int zd = tok / 49, rem = tok % 49, yh = rem / 7, xp = rem % 7;
  int d = wd * 2 + zd, h = wh * 7 + yh, w = wwj * 7 + xp;     // shifted coords
  int od = (d + 1) & 7, oh = (h + 3) % 28, ow = (w + 3) % 28; // inverse roll
  const float* row = x + (size_t)(((bidx * 8 + od) * 28 + oh) * 28 + ow) * Cc;
  float v[12]; float s = 0.f;
#pragma unroll
  for (int i = 0; i < 12; ++i) { v[i] = row[lane + 32 * i]; s += v[i]; }
  s = wave_sum(s);
  float mean = s * (1.0f / Cc);
  float q = 0.f;
#pragma unroll
  for (int i = 0; i < 12; ++i) { float dv = v[i] - mean; q += dv * dv; }
  q = wave_sum(q);
  float inv = rsqrtf(q * (1.0f / Cc) + 1e-5f);
#pragma unroll
  for (int i = 0; i < 12; ++i) {
    int j = lane + 32 * i;
    orow[j] = f2b((v[i] - mean) * inv * g[j] + bta[j]);
  }
}

// ---------------------------------------------------------------------------
// Generic WMMA GEMM: C[M][N] = A[M][K] * W[N][K]^T + bias, epilogue variants.
// Block 256 thr (8 waves). Tile BM=256, BN=64; per wave: 2 M-tiles x 4 N-tiles
// (8 WMMAs per 12 LDS b128 reads per K-chunk of 32).
// EP: 0=f32, 1=bf16, 2=gelu->bf16, 3=relu->bf16, 4=qkv scatter,
//     5=add x1 -> f32 out, 6=accumulate into f32 out
// ---------------------------------------------------------------------------
template <int EP>
__global__ __launch_bounds__(256)
void gemm_bf16(const bf16* __restrict__ A, const bf16* __restrict__ W,
               const float* __restrict__ bias, void* __restrict__ out,
               const float* __restrict__ x1, int M, int N, int K, int ldo) {
  __shared__ __align__(16) bf16 sA[256 * 32];   // 16 KB
  __shared__ __align__(16) bf16 sW[64 * 32];    //  4 KB
  const int bm = blockIdx.x * 256;
  const int bn = blockIdx.y * 64;
  const int tid = threadIdx.x;
  const int wid = tid >> 5, lane = tid & 31;
  const int lm = lane & 15, hi = lane >> 4;
  v8f acc[2][4] = {};

  for (int kc = 0; kc < K; kc += 32) {
    __syncthreads();
#pragma unroll
    for (int i = 0; i < 4; ++i) {               // A tile: 256x32 bf16
      int c = tid + i * 256;
      int r = c >> 2, qtr = c & 3;
      *(uint4*)&sA[r * 32 + qtr * 8] =
          *(const uint4*)&A[(size_t)(bm + r) * K + kc + qtr * 8];
    }
    {                                           // W tile: 64x32 bf16
      int r = tid >> 2, qtr = tid & 3;
      *(uint4*)&sW[r * 32 + qtr * 8] =
          *(const uint4*)&W[(size_t)(bn + r) * K + kc + qtr * 8];
    }
    __syncthreads();
    BF16Tile a[2];
#pragma unroll
    for (int mt = 0; mt < 2; ++mt) {
      const bf16* ap = &sA[((wid * 2 + mt) * 16 + lm) * 32];
      a[mt].q[0] = *(const uint4*)(ap + hi * 8);
      a[mt].q[1] = *(const uint4*)(ap + 16 + hi * 8);
    }
#pragma unroll
    for (int nt = 0; nt < 4; ++nt) {
      BF16Tile b;
      const bf16* bp = &sW[(nt * 16 + lm) * 32];
      b.q[0] = *(const uint4*)(bp + hi * 8);
      b.q[1] = *(const uint4*)(bp + 16 + hi * 8);
#pragma unroll
      for (int mt = 0; mt < 2; ++mt) {
        acc[mt][nt] = __builtin_amdgcn_wmma_f32_16x16x32_bf16(
            false, a[mt].v, false, b.v, (short)0, acc[mt][nt], false, false);
      }
    }
  }

  const float qscale = 0.17677669529663687f;    // 32^-0.5
#pragma unroll
  for (int mt = 0; mt < 2; ++mt) {
#pragma unroll
    for (int nt = 0; nt < 4; ++nt) {
      int n = bn + nt * 16 + lm;
      float bv = bias[n];
#pragma unroll
      for (int r = 0; r < 8; ++r) {
        int m = bm + (wid * 2 + mt) * 16 + r + 8 * hi;
        float v = acc[mt][nt][r] + bv;
        if constexpr (EP == 0) {
          ((float*)out)[(size_t)m * ldo + n] = v;
        } else if constexpr (EP == 1) {
          ((bf16*)out)[(size_t)m * ldo + n] = f2b(v);
        } else if constexpr (EP == 2) {
          float gl = 0.5f * v * (1.0f + erff(v * 0.70710678118654752f));
          ((bf16*)out)[(size_t)m * ldo + n] = f2b(gl);
        } else if constexpr (EP == 3) {
          ((bf16*)out)[(size_t)m * ldo + n] = f2b(fmaxf(v, 0.0f));
        } else if constexpr (EP == 4) {
          int win = m / NPAD, tok = m % NPAD;
          int which = n / Cc, rn = n % Cc, head = rn >> 5, dd = rn & 31;
          bf16* base = (bf16*)out + ((size_t)(win * 3 + which) * HEADS + head) * (NPAD * HD);
          if (which == 0)      base[tok * HD + dd] = f2b(v * qscale);   // Q [tok][hd]
          else if (which == 1) base[tok * HD + dd] = f2b(v);            // K [tok][hd]
          else                 base[dd * NPAD + tok] = f2b(v);          // V^T [hd][tok]
        } else if constexpr (EP == 5) {
          ((float*)out)[(size_t)m * ldo + n] = v + x1[(size_t)m * ldo + n];
        } else {
          float* o = (float*)out + (size_t)m * ldo + n;
          *o = *o + v;
        }
      }
    }
  }
}

// ---------------------------------------------------------------------------
// Windowed attention: one block (7 waves) per (window, head).
// S = Q K^T + biasT + maskT; softmax; O = P V.
// ---------------------------------------------------------------------------
__global__ __launch_bounds__(224)
void attn_kernel(const bf16* __restrict__ qkv, const float* __restrict__ biasT,
                 const float* __restrict__ maskT, bf16* __restrict__ aout) {
  __shared__ __align__(16) float s_s[NSQ];     // 50176 B, later aliased as bf16 P
  const int blk = blockIdx.x;
  const int win = blk / HEADS, head = blk % HEADS;
  const int iw = win & 63;
  const int tid = threadIdx.x, wid = tid >> 5, lane = tid & 31;
  const int lm = lane & 15, hi = lane >> 4;
  const bf16* qp = qkv + ((size_t)(win * 3 + 0) * HEADS + head) * (NPAD * HD);
  const bf16* kp = qkv + ((size_t)(win * 3 + 1) * HEADS + head) * (NPAD * HD);
  const bf16* vp = qkv + ((size_t)(win * 3 + 2) * HEADS + head) * (NPAD * HD);
  const float* bT = biasT + (size_t)head * NSQ;
  const float* mT = maskT + (size_t)iw * NSQ;

  // ---- phase 1: S = Q K^T + bias + mask -> LDS f32 ----
  {
    BF16Tile a;
    const bf16* ap = qp + (wid * 16 + lm) * HD;
    a.q[0] = *(const uint4*)(ap + hi * 8);
    a.q[1] = *(const uint4*)(ap + 16 + hi * 8);
#pragma unroll 1
    for (int nt = 0; nt < 7; ++nt) {
      BF16Tile b;
      const bf16* bp = kp + (nt * 16 + lm) * HD;
      b.q[0] = *(const uint4*)(bp + hi * 8);
      b.q[1] = *(const uint4*)(bp + 16 + hi * 8);
      v8f c = {};
      c = __builtin_amdgcn_wmma_f32_16x16x32_bf16(
          false, a.v, false, b.v, (short)0, c, false, false);
      int n = nt * 16 + lm;
#pragma unroll
      for (int r = 0; r < 8; ++r) {
        int m = wid * 16 + r + 8 * hi;
        int e = m * NPAD + n;
        s_s[e] = c[r] + bT[e] + mT[e];
      }
    }
  }
  __syncthreads();

  // ---- phase 2: row softmax, in-place bf16 P (row stride 224 halves) ----
  if (tid < NPAD) {
    float* row = &s_s[tid * NPAD];
    float mx = -3.4e38f;
    for (int j = 0; j < NPAD; ++j) mx = fmaxf(mx, row[j]);
    float sum = 0.f;
    for (int j = 0; j < NPAD; ++j) sum += __expf(row[j] - mx);
    float inv = 1.0f / sum;
    bf16* prow = (bf16*)row;                  // aliases this row's own footprint
    for (int j = 0; j < 128; ++j) {
      float p = (j < NPAD) ? __expf(row[j] - mx) * inv : 0.0f;
      prow[j] = f2b(p);                       // overwrites only floats < j (consumed)
    }
  }
  __syncthreads();

  // ---- phase 3: O = P V  (K padded to 128) ----
  {
    const bf16* pl = (const bf16*)s_s;        // row stride 224 halves
    v8f o[2] = {};
#pragma unroll 1
    for (int kc = 0; kc < 4; ++kc) {
      BF16Tile a;
      const bf16* ap = pl + (wid * 16 + lm) * 224 + kc * 32;
      a.q[0] = *(const uint4*)(ap + hi * 8);
      a.q[1] = *(const uint4*)(ap + 16 + hi * 8);
#pragma unroll
      for (int nt = 0; nt < 2; ++nt) {
        BF16Tile b;                            // V^T rows: [hd][tok]
        const bf16* bp = vp + (nt * 16 + lm) * NPAD + kc * 32;
        b.q[0] = *(const uint4*)(bp + hi * 8);
        b.q[1] = *(const uint4*)(bp + 16 + hi * 8);
        o[nt] = __builtin_amdgcn_wmma_f32_16x16x32_bf16(
            false, a.v, false, b.v, (short)0, o[nt], false, false);
      }
    }
#pragma unroll
    for (int nt = 0; nt < 2; ++nt) {
      int col = head * HD + nt * 16 + lm;
#pragma unroll
      for (int r = 0; r < 8; ++r) {
        int m = wid * 16 + r + 8 * hi;
        aout[(size_t)(win * NPAD + m) * Cc + col] = f2b(o[nt][r]);
      }
    }
  }
}

// ---------------------------------------------------------------------------
// window reverse + roll(+1,+3,+3) + residual: x1 = x + scatter(proj_out)
// ---------------------------------------------------------------------------
__global__ __launch_bounds__(256)
void unshift_add_kernel(const float* __restrict__ x, const float* __restrict__ proj,
                        float* __restrict__ x1) {
  size_t idx = (size_t)blockIdx.x * 256 + threadIdx.x;   // < 50176*384
  int c = (int)(idx % Cc);
  size_t t = idx / Cc;
  int b = (int)(t / 6272); int r = (int)(t % 6272);
  int d = r / 784; r %= 784; int h = r / 28; int w = r % 28;
  int d2 = (d + 7) & 7, h2 = (h + 25) % 28, w2 = (w + 25) % 28;  // forward roll
  int win = b * 64 + (d2 >> 1) * 16 + (h2 / 7) * 4 + (w2 / 7);
  int tok = (d2 & 1) * 49 + (h2 % 7) * 7 + (w2 % 7);
  x1[idx] = x[idx] + proj[(size_t)(win * NPAD + tok) * Cc + c];
}

// ---------------------------------------------------------------------------
// LN2 (-> y bf16) and adapter LN of y (-> adln bf16). One wave per token.
// ---------------------------------------------------------------------------
__global__ __launch_bounds__(256)
void ln2_kernel(const float* __restrict__ x1,
                const float* __restrict__ n2g, const float* __restrict__ n2b,
                const float* __restrict__ adg, const float* __restrict__ adb,
                bf16* __restrict__ y, bf16* __restrict__ adln) {
  int wid = threadIdx.x >> 5, lane = threadIdx.x & 31;
  int t = blockIdx.x * 8 + wid;               // 0 .. 50175
  const float* row = x1 + (size_t)t * Cc;
  float v[12]; float s = 0.f;
#pragma unroll
  for (int i = 0; i < 12; ++i) { v[i] = row[lane + 32 * i]; s += v[i]; }
  s = wave_sum(s);
  float mean = s * (1.0f / Cc);
  float q = 0.f;
#pragma unroll
  for (int i = 0; i < 12; ++i) { float dv = v[i] - mean; q += dv * dv; }
  q = wave_sum(q);
  float inv = rsqrtf(q * (1.0f / Cc) + 1e-5f);
  float yv[12]; float s2 = 0.f;
#pragma unroll
  for (int i = 0; i < 12; ++i) {
    int j = lane + 32 * i;
    yv[i] = (v[i] - mean) * inv * n2g[j] + n2b[j];
    y[(size_t)t * Cc + j] = f2b(yv[i]);
    s2 += yv[i];
  }
  s2 = wave_sum(s2);
  float mean2 = s2 * (1.0f / Cc);
  float q2 = 0.f;
#pragma unroll
  for (int i = 0; i < 12; ++i) { float dv = yv[i] - mean2; q2 += dv * dv; }
  q2 = wave_sum(q2);
  float inv2 = rsqrtf(q2 * (1.0f / Cc) + 1e-5f);
#pragma unroll
  for (int i = 0; i < 12; ++i) {
    int j = lane + 32 * i;
    adln[(size_t)t * Cc + j] = f2b((yv[i] - mean2) * inv2 * adg[j] + adb[j]);
  }
}

// ---------------------------------------------------------------------------
// Host: workspace layout (phase-reused; peak ~409 MB) + launch sequence
// ---------------------------------------------------------------------------
extern "C" void kernel_launch(void* const* d_in, const int* in_sizes, int n_in,
                              void* d_out, int out_size, void* d_ws, size_t ws_size,
                              hipStream_t stream) {
  (void)in_sizes; (void)n_in; (void)out_size; (void)ws_size;
  const float* x      = (const float*)d_in[0];
  const float* n1g    = (const float*)d_in[1];
  const float* n1b    = (const float*)d_in[2];
  const float* qkv_w  = (const float*)d_in[3];
  const float* qkv_b  = (const float*)d_in[4];
  const float* rpb    = (const float*)d_in[5];
  const float* proj_w = (const float*)d_in[6];
  const float* proj_b = (const float*)d_in[7];
  const float* n2g    = (const float*)d_in[8];
  const float* n2b    = (const float*)d_in[9];
  const float* fc1_w  = (const float*)d_in[10];
  const float* fc1_b  = (const float*)d_in[11];
  const float* fc2_w  = (const float*)d_in[12];
  const float* fc2_b  = (const float*)d_in[13];
  const float* adlng  = (const float*)d_in[14];
  const float* adlnb  = (const float*)d_in[15];
  const float* addn_w = (const float*)d_in[16];
  const float* addn_b = (const float*)d_in[17];
  const float* adup_w = (const float*)d_in[18];
  const float* adup_b = (const float*)d_in[19];
  float* out = (float*)d_out;

  char* ws = (char*)d_ws;
  // weights (bf16), persistent through the call
  bf16* wqkv  = (bf16*)(ws + 0);                       // 1152*384
  bf16* wproj = (bf16*)(ws + 884736);                  // 384*384
  bf16* wfc1  = (bf16*)(ws + 1179648);                 // 1536*384
  bf16* wfc2  = (bf16*)(ws + 2359296);                 // 384*1536
  bf16* wdn   = (bf16*)(ws + 3538944);                 // 64*384
  bf16* wup   = (bf16*)(ws + 3588096);                 // 384*64
  // attention tables (persistent through attention)
  float* biasT = (float*)(ws + 3637248);               // 12*12544 f32  (602112 B)
  float* maskT = (float*)(ws + 4239360);               // 64*12544 f32 (3211264 B) end 7450624
  // activations (phase reuse)
  bf16*  xw    = (bf16*)(ws + (8ull << 20));           // [57344][384] bf16
  bf16*  qkvb  = (bf16*)(ws + (56ull << 20));          // [512][3][12][112*32] bf16
  bf16*  aoutb = (bf16*)(ws + (192ull << 20));         // [57344][384] bf16
  float* projo = (float*)(ws + (240ull << 20));        // [57344][384] f32
  float* x1    = (float*)(ws + (332ull << 20));        // [50176][384] f32
  bf16*  ybuf  = (bf16*)(ws + (8ull << 20));           // reuse xw region
  bf16*  adln  = (bf16*)(ws + (56ull << 20));          // reuse qkv region
  bf16*  mbuf  = (bf16*)(ws + (100ull << 20));         // [50176][1536] bf16 (reuse)
  bf16*  addwn = (bf16*)(ws + (264ull << 20));         // [50176][64] bf16 (reuse)

  // 1) weights -> bf16; attention tables
  auto cvt = [&](const float* s, bf16* d, int n) {
    f2bf_kernel<<<(n + 255) / 256, 256, 0, stream>>>(s, d, n);
  };
  cvt(qkv_w,  wqkv,  3 * Cc * Cc);
  cvt(proj_w, wproj, Cc * Cc);
  cvt(fc1_w,  wfc1,  MLPH * Cc);
  cvt(fc2_w,  wfc2,  Cc * MLPH);
  cvt(addn_w, wdn,   BOT * Cc);
  cvt(adup_w, wup,   Cc * BOT);
  build_bias_kernel<<<(HEADS * NSQ) / 256, 256, 0, stream>>>(rpb, biasT);
  build_mask_kernel<<<(64 * NSQ) / 256, 256, 0, stream>>>(maskT);

  // 2) LN1 + shift + partition
  ln1_part_kernel<<<M_WIN / 8, 256, 0, stream>>>(x, n1g, n1b, xw);

  // 3) QKV GEMM (scatter epilogue: Q scaled, V transposed)
  gemm_bf16<4><<<dim3(M_WIN / 256, (3 * Cc) / 64), 256, 0, stream>>>(
      xw, wqkv, qkv_b, (void*)qkvb, nullptr, M_WIN, 3 * Cc, Cc, 0);

  // 4) attention per (window, head)
  attn_kernel<<<NWIN * HEADS, 224, 0, stream>>>(qkvb, biasT, maskT, aoutb);

  // 5) output projection (f32)
  gemm_bf16<0><<<dim3(M_WIN / 256, Cc / 64), 256, 0, stream>>>(
      aoutb, wproj, proj_b, (void*)projo, nullptr, M_WIN, Cc, Cc, Cc);

  // 6) window reverse + unshift + residual
  unshift_add_kernel<<<(M_TOK * Cc) / 256, 256, 0, stream>>>(x, projo, x1);

  // 7) LN2 + adapter LN
  ln2_kernel<<<M_TOK / 8, 256, 0, stream>>>(x1, n2g, n2b, adlng, adlnb, ybuf, adln);

  // 8) fc1 + GELU
  gemm_bf16<2><<<dim3(M_TOK / 256, MLPH / 64), 256, 0, stream>>>(
      ybuf, wfc1, fc1_b, (void*)mbuf, nullptr, M_TOK, MLPH, Cc, MLPH);

  // 9) adapter down + ReLU
  gemm_bf16<3><<<dim3(M_TOK / 256, BOT / 64), 256, 0, stream>>>(
      adln, wdn, addn_b, (void*)addwn, nullptr, M_TOK, BOT, Cc, BOT);

  // 10) fc2, fused with x1 residual -> d_out
  gemm_bf16<5><<<dim3(M_TOK / 256, Cc / 64), 256, 0, stream>>>(
      mbuf, wfc2, fc2_b, (void*)out, x1, M_TOK, Cc, MLPH, Cc);

  // 11) adapter up accumulates into d_out
  gemm_bf16<6><<<dim3(M_TOK / 256, Cc / 64), 256, 0, stream>>>(
      addwn, wup, adup_b, (void*)out, nullptr, M_TOK, Cc, BOT, Cc);
}
